// LstmLayer_47734266527944
// MI455X (gfx1250) — matrix operand order repaired and checked by reading, not compiled, not run on hardware
//
#include <hip/hip_runtime.h>
#include <hip/hip_bf16.h>
#include <stdint.h>

// ---------------------------------------------------------------------------
// LSTM on MI455X (gfx1250), wave32 + WMMA, persistent recurrent kernel.
//   gates[m][n] = sum_k A[m][k] * Wall[k][n],  A = [x_t | h_{t-1}]  (64 x 1536)
//   Wall = [Wcat; Ucat] (1536 x 4096), n = gate*1024 + hidden_col
// v_wmma_f32_16x16x32_f16, weights LDS-resident (192KB), split-K across wave
// pairs (2 waves/SIMD), double-buffered B fragments, cell state VGPR-resident,
// next-step x warmed via global_prefetch during the inter-step barrier.
// ---------------------------------------------------------------------------

typedef __attribute__((ext_vector_type(16))) _Float16 v16h;
typedef __attribute__((ext_vector_type(8)))  _Float16 v8h;
typedef __attribute__((ext_vector_type(8)))  float    v8f;

#define BATCH 64
#define SEQ   512
#define DIN   512
#define HID   1024
#define KC_X  16                     // 32-wide K chunks from x
#define KC_TOT 48
#define KC_HALF 24                   // chunks per wave (split-K)
#define GRID  64                     // one block per hidden col-tile
#define BLOCK 256                    // 8 waves: [m-tile 0..3] x [K half 0..1]
#define GATE_LDS_STRIDE (48 * 1024)  // 48 fragments * 1KB per gate
#define RED_OFF  (4 * GATE_LDS_STRIDE)      // 192 KB: reduce buffer start
#define LDS_BYTES (RED_OFF + 16 * 1024)     // 208 KB total

// ---- workspace layout (bytes) ----
#define WALL_OFF   0ull
#define WALL_BYTES (256ull * 48 * 512 * 2)           // 12 MB swizzled f16 weights
#define XT_OFF     (WALL_OFF + WALL_BYTES)
#define XT_BYTES   (512ull * 64 * 512 * 2)           // 32 MB time-major f16 x
#define BIAS_OFF   (XT_OFF + XT_BYTES)
#define BIAS_BYTES (4096ull * 4)
#define HB_OFF     (BIAS_OFF + BIAS_BYTES)
#define HB_BYTES   (2ull * 64 * 1024 * 2)            // ping-pong h (f16)
#define CNT_OFF    (HB_OFF + HB_BYTES)

#define WALL_N 6291456u    // 256*48*512 halves
#define XT_N   16777216u   // 512*64*512 halves
#define PREP_TOTAL (WALL_N + XT_N + 4096u + 131072u + 1u)

// ---------------------------------------------------------------------------
// Prep: swizzle weights into WMMA B-fragment order, x -> time-major f16,
// concat biases, zero h buffers + barrier counter.
//   B fragment (32x16 f16): lane<16 -> col n=lane, half j -> K=j;
//   lanes 16-31 -> col n=lane-16, half j -> K=16+j.  Fragment = 1KB,
//   lane-major: half index = lane*16 + j.
// ---------------------------------------------------------------------------
__global__ void lstm_prep_kernel(
    const float* __restrict__ x,
    const float* __restrict__ Wi, const float* __restrict__ Ui, const float* __restrict__ bi,
    const float* __restrict__ Wf, const float* __restrict__ Uf, const float* __restrict__ bf,
    const float* __restrict__ Wc, const float* __restrict__ Uc, const float* __restrict__ bc,
    const float* __restrict__ Wo, const float* __restrict__ Uo, const float* __restrict__ bo,
    char* __restrict__ ws)
{
    unsigned i = blockIdx.x * blockDim.x + threadIdx.x;
    if (i >= PREP_TOTAL) return;

    _Float16* wall = (_Float16*)(ws + WALL_OFF);
    _Float16* xt   = (_Float16*)(ws + XT_OFF);
    float*    bias = (float*)(ws + BIAS_OFF);
    _Float16* hbuf = (_Float16*)(ws + HB_OFF);
    unsigned* cnt  = (unsigned*)(ws + CNT_OFF);

    if (i < WALL_N) {
        unsigned e   = i;
        unsigned nt  = e / 24576u;            // 48*512
        unsigned rem = e % 24576u;
        unsigned kc  = rem >> 9;
        unsigned p   = rem & 511u;
        unsigned l   = p >> 4;                // lane 0..31
        unsigned j   = p & 15u;
        unsigned K   = kc * 32u + ((l >= 16u) ? (16u + j) : j);
        unsigned n   = nt * 16u + (l & 15u);
        unsigned g   = n >> 10;               // gate 0..3 (i,f,c,o)
        unsigned c   = n & 1023u;
        const float* Wp = (g == 0) ? Wi : (g == 1) ? Wf : (g == 2) ? Wc : Wo;
        const float* Up = (g == 0) ? Ui : (g == 1) ? Uf : (g == 2) ? Uc : Uo;
        float v = (K < 512u) ? Wp[(size_t)K * 1024u + c]
                             : Up[(size_t)(K - 512u) * 1024u + c];
        wall[e] = (_Float16)v;
    } else if (i < WALL_N + XT_N) {
        unsigned idx = i - WALL_N;
        unsigned t   = idx / 32768u;          // 64*512
        unsigned rem = idx & 32767u;
        unsigned m   = rem >> 9;
        unsigned k   = rem & 511u;
        xt[idx] = (_Float16)x[((size_t)m * SEQ + t) * DIN + k];
    } else if (i < WALL_N + XT_N + 4096u) {
        unsigned n = i - (WALL_N + XT_N);
        unsigned g = n >> 10, c = n & 1023u;
        const float* bp = (g == 0) ? bi : (g == 1) ? bf : (g == 2) ? bc : bo;
        bias[n] = bp[c];
    } else if (i < WALL_N + XT_N + 4096u + 131072u) {
        hbuf[i - (WALL_N + XT_N + 4096u)] = (_Float16)0.0f;
    } else {
        *cnt = 0u;
    }
}

// ---------------------------------------------------------------------------
__device__ __forceinline__ v8f wmma_f16(v16h a, v16h b, v8f c) {
    return __builtin_amdgcn_wmma_f32_16x16x32_f16(
        false, a, false, b, (short)0, c, false, false);
}
__device__ __forceinline__ float sigm_f(float x) {
    return 1.0f / (1.0f + __expf(-x));
}
__device__ __forceinline__ float tanh_f(float x) {
    return 1.0f - 2.0f / (__expf(2.0f * x) + 1.0f);   // saturates for |x| large
}
__device__ __forceinline__ v16h load_a16(const _Float16* p) {
    return *(const v16h*)(p);
}

// load all 4 gate B-fragments of one K-chunk from LDS (8x ds_load_b128)
#define LOAD_B(dst, kcConst)                                                   \
    { _Pragma("unroll")                                                        \
      for (int g = 0; g < 4; ++g)                                              \
          dst[g] = *(const v16h*)(smem + ((kcConst) << 10) +                   \
                                  g * GATE_LDS_STRIDE + (lane << 5)); }

#define DO_WMMA4(aC, bArr)                                                     \
    { _Pragma("unroll")                                                        \
      for (int g = 0; g < 4; ++g) acc[g] = wmma_f16((aC), bArr[g], acc[g]); }

__global__ __launch_bounds__(BLOCK, 1)
void lstm_recurrent_kernel(char* __restrict__ ws, float* __restrict__ out)
{
    extern __shared__ char smem[];   // 192 KB B fragments + 16 KB reduce

    const _Float16* wall = (const _Float16*)(ws + WALL_OFF);
    const _Float16* xt   = (const _Float16*)(ws + XT_OFF);
    const float*    bias = (const float*)(ws + BIAS_OFF);
    _Float16*       hbuf = (_Float16*)(ws + HB_OFF);
    unsigned*       cnt  = (unsigned*)(ws + CNT_OFF);

    const int tid   = threadIdx.x;
    const int wave  = tid >> 5;          // 0..7
    const int kwav  = wave & 3;          // M tile 0..3
    const int khalf = wave >> 2;         // 0: K chunks 0..23, 1: 24..47
    const int lane  = tid & 31;
    const int lrow  = lane & 15;
    const int lhalf = lane >> 4;
    const int ct    = blockIdx.x;        // hidden col tile 0..63

    // ---- stage B slice (4 gates x 48 fragments) into LDS, once ----
    for (int g = 0; g < 4; ++g) {
        const uint4* src = (const uint4*)(wall + (size_t)(g * 64 + ct) * 48 * 512);
        uint4* dst = (uint4*)(smem + g * GATE_LDS_STRIDE);
        for (int q = tid; q < GATE_LDS_STRIDE / 16; q += BLOCK) dst[q] = src[q];
    }
    __syncthreads();

    const int m0   = kwav * 16;
    const int arow = m0 + lrow;              // row this lane supplies for A
    const int ccol = ct * 16 + lrow;         // hidden col this lane owns in C/D
    const float bI = bias[0 * HID + ccol];
    const float bF = bias[1 * HID + ccol];
    const float bG = bias[2 * HID + ccol];
    const float bO = bias[3 * HID + ccol];

    v8f creg = {};                           // cell state, register-resident

    for (int t = 0; t < SEQ; ++t) {
        const _Float16* hprev = hbuf + (size_t)(t & 1) * (BATCH * HID);
        _Float16*       hnext = hbuf + (size_t)((t + 1) & 1) * (BATCH * HID);

        v8f acc[4] = {{}, {}, {}, {}};

        const _Float16* xrow = xt + ((size_t)t * BATCH + arow) * DIN + lhalf * 16;
        const _Float16* hrow = hprev + (size_t)arow * HID + lhalf * 16;

        if (khalf == 0) {
            // kc 0..15 from x, 16..23 from h; depth-2 A pipeline +
            // double-buffered B (next chunk's 4 fragments prefetched)
            v16h a0 = load_a16(xrow);
            v16h a1 = load_a16(xrow + 32);
            v16h bbuf[2][4];
            LOAD_B(bbuf[0], 0);
            #pragma unroll
            for (int k = 0; k < 24; ++k) {
                const int cur = k & 1, nxt = cur ^ 1;
                if (k < 23) LOAD_B(bbuf[nxt], k + 1);
                v16h aC = a0;
                a0 = a1;
                if (k < 14)      a1 = load_a16(xrow + (k + 2) * 32);
                else if (k < 22) a1 = load_a16(hrow + (k - 14) * 32);
                DO_WMMA4(aC, bbuf[cur]);
            }
        } else {
            // kc 24..47, all from h (chunks 8..31)
            const _Float16* ah = hrow + 8 * 32;
            v16h a0 = load_a16(ah);
            v16h a1 = load_a16(ah + 32);
            v16h bbuf[2][4];
            LOAD_B(bbuf[0], 24);
            #pragma unroll
            for (int k = 0; k < 24; ++k) {
                const int cur = k & 1, nxt = cur ^ 1;
                if (k < 23) LOAD_B(bbuf[nxt], 24 + k + 1);
                v16h aC = a0;
                a0 = a1;
                if (k < 22) a1 = load_a16(ah + (k + 2) * 32);
                DO_WMMA4(aC, bbuf[cur]);
            }
        }

        if (khalf) {
            // high-K waves publish partial sums
            #pragma unroll
            for (int g = 0; g < 4; ++g)
                *(v8f*)(smem + RED_OFF + ((kwav * 4 + g) << 10) + (lane << 5))
                    = acc[g];
        }
        __syncthreads();

        if (!khalf) {
            #pragma unroll
            for (int g = 0; g < 4; ++g)
                acc[g] += *(const v8f*)(smem + RED_OFF +
                                        ((kwav * 4 + g) << 10) + (lane << 5));
            // ---- pointwise gates; comp r -> row m0 + lhalf*8 + r ----
            #pragma unroll
            for (int r = 0; r < 8; ++r) {
                float iv = sigm_f(acc[0][r] + bI);
                float fv = sigm_f(acc[1][r] + bF);
                float gv = tanh_f(acc[2][r] + bG);
                float ov = sigm_f(acc[3][r] + bO);
                float cv = fv * creg[r] + iv * gv;
                creg[r] = cv;
                float hv = ov * tanh_f(cv);
                const int mrow = m0 + lhalf * 8 + r;
                out[((size_t)mrow * SEQ + t) * HID + ccol] = ov;       // o_t
                hnext[(size_t)mrow * HID + ccol] = (_Float16)hv;
                if (t == SEQ - 1)
                    out[(size_t)BATCH * SEQ * HID + (size_t)mrow * HID + ccol] = hv;
            }
            // warm next step's x slice (16 rows * 1KB, lane-distributed lines)
            if (t + 1 < SEQ) {
                const char* xn = (const char*)(xt +
                    ((size_t)(t + 1) * BATCH + m0) * DIN);
                #pragma unroll
                for (int q = 0; q < 4; ++q)
                    __builtin_prefetch(xn + (q * 32 + lane) * 128, 0, 1);
            }
        }

        // ---- grid-wide barrier (monotonic counter, agent-scope fences) ----
        __builtin_amdgcn_fence(__ATOMIC_RELEASE, "agent");
        __syncthreads();
        if (tid == 0) {
            atomicAdd(cnt, 1u);
            const unsigned target = (unsigned)(t + 1) * (unsigned)GRID;
            while (__hip_atomic_load(cnt, __ATOMIC_RELAXED,
                                     __HIP_MEMORY_SCOPE_AGENT) < target)
                __builtin_amdgcn_s_sleep(2);
        }
        __syncthreads();
        __builtin_amdgcn_fence(__ATOMIC_ACQUIRE, "agent");  // invalidate L0: fresh h
    }
}

// ---------------------------------------------------------------------------
extern "C" void kernel_launch(void* const* d_in, const int* in_sizes, int n_in,
                              void* d_out, int out_size, void* d_ws, size_t ws_size,
                              hipStream_t stream) {
    const float* x  = (const float*)d_in[0];
    const float* Wi = (const float*)d_in[1];
    const float* Ui = (const float*)d_in[2];
    const float* bi = (const float*)d_in[3];
    const float* Wf = (const float*)d_in[4];
    const float* Uf = (const float*)d_in[5];
    const float* bf = (const float*)d_in[6];
    const float* Wc = (const float*)d_in[7];
    const float* Uc = (const float*)d_in[8];
    const float* bc = (const float*)d_in[9];
    const float* Wo = (const float*)d_in[10];
    const float* Uo = (const float*)d_in[11];
    const float* bo = (const float*)d_in[12];

    char*  ws  = (char*)d_ws;
    float* out = (float*)d_out;

    const unsigned prep_blocks = (PREP_TOTAL + 255u) / 256u;
    lstm_prep_kernel<<<dim3(prep_blocks), dim3(256), 0, stream>>>(
        x, Wi, Ui, bi, Wf, Uf, bf, Wc, Uc, bc, Wo, Uo, bo, ws);

    lstm_recurrent_kernel<<<dim3(GRID), dim3(BLOCK), LDS_BYTES, stream>>>(ws, out);
}
